// PointerNetwork_2379411882354
// MI455X (gfx1250) — matrix operand order repaired and checked by reading, not compiled, not run on hardware
//
#include <hip/hip_runtime.h>
#include <hip/hip_bf16.h>

// ---------------------------------------------------------------------------
// PointerNetwork forward on gfx1250 (CDNA5, wave32, WMMA, async LDS copies).
// B=128, S=128, IN=16, D=256, H=256.
// All WMMA A/B operands are kept in f16 end-to-end (producers emit f16),
// accumulation and nonlinearities in f32.
// ---------------------------------------------------------------------------

typedef __attribute__((ext_vector_type(16))) _Float16 hv16;
typedef __attribute__((ext_vector_type(8)))  _Float16 hv8;
typedef __attribute__((ext_vector_type(8)))  float    fv8;

#define BB 128
#define SS 128
#define IN 16
#define DD 256
#define HH 256

union HPack { hv16 v; hv8 h[2]; };

// A fragment (16x32 f16, MxK) from a row-major f16 buffer (global or LDS).
// ISA layout: lanes 0-15 / 16-31 both carry M=0..15;
// elems 0..7  -> K = k0 + 8*half + e      (contiguous 8 halfs, 16B)
// elems 8..15 -> K = k0 + 16 + 8*half + e (contiguous 8 halfs, 16B)
__device__ __forceinline__ hv16 frag_a_h(const _Float16* A, int lda, int m0, int k0, int lane) {
  int row = m0 + (lane & 15);
  int hl  = lane >> 4;
  const _Float16* p = A + (size_t)row * lda + k0 + hl * 8;
  HPack u;
  u.h[0] = *(const hv8*)p;
  u.h[1] = *(const hv8*)(p + 16);
  return u.v;
}

// B fragment (32x16 f16, KxN) for C = A * W^T with W row-major f16 (N x K):
// B[k][n] = W[n][k]. elems 0..7 -> K = k0 + 8*half + e, elems 8..15 -> K = k0 + 16 + 8*half + e,
// N = lane&15 -> 2 x 8 contiguous halfs per lane (2 x 16B).
__device__ __forceinline__ hv16 frag_b_wT(const _Float16* W, int ldw, int n0, int k0, int lane) {
  int col = n0 + (lane & 15);
  int hl  = lane >> 4;
  const _Float16* p = W + (size_t)col * ldw + k0 + hl * 8;
  HPack u;
  u.h[0] = *(const hv8*)p;
  u.h[1] = *(const hv8*)(p + 16);
  return u.v;
}

#define WMMA(a, b, c) __builtin_amdgcn_wmma_f32_16x16x32_f16( \
    false, (a), false, (b), (short)0, (c), false, false)

__device__ __forceinline__ float sigm(float x) { return 1.0f / (1.0f + __expf(-x)); }

// --- CDNA5 async global->LDS copy (ASYNCcnt), 16 bytes per lane -------------
// LDS byte address = low 32 bits of the generic address (aperture rule).
__device__ __forceinline__ void async_copy_b128(unsigned lds_addr, const void* gptr) {
  asm volatile("global_load_async_to_lds_b128 %0, %1, off"
               :: "v"(lds_addr), "v"((unsigned long long)(size_t)gptr)
               : "memory");
}
__device__ __forceinline__ void wait_async_all() {
  asm volatile("s_wait_asynccnt 0x0" ::: "memory");
}

// ---------------------------------------------------------------------------
// Generic GEMM: C[M x N] = act(A_f16[M x K] @ W_f16[N x K]^T + bias).
// grid = (N/16, M/16), block = 32 (one wave per 16x16 tile). Pure b128+WMMA.
// ---------------------------------------------------------------------------
__global__ void gemm_wmma_kernel(const _Float16* __restrict__ A, const _Float16* __restrict__ W,
                                 const float* __restrict__ bias, void* __restrict__ Cout,
                                 int K, int lda, int ldw, int ldc, int act, int half_out) {
  int lane = threadIdx.x & 31;
  int n0 = blockIdx.x * 16;
  int m0 = blockIdx.y * 16;
  fv8 acc = {};
  for (int k0 = 0; k0 < K; k0 += 32) {
    if (k0 + 32 < K)  // keep the weight stream ahead in L2 (global_prefetch_b8)
      __builtin_prefetch(W + (size_t)(n0 + (lane & 15)) * ldw + k0 + 32, 0, 1);
    hv16 a = frag_a_h(A, lda, m0, k0, lane);
    hv16 b = frag_b_wT(W, ldw, n0, k0, lane);
    acc = WMMA(a, b, acc);
  }
  int n = lane & 15, hl = lane >> 4;
  float bv = bias ? bias[n0 + n] : 0.0f;
  for (int r = 0; r < 8; ++r) {
    float v = acc[r] + bv;
    if (act) v = tanhf(v);
    size_t idx = (size_t)(m0 + r + 8 * hl) * ldc + n0 + n;
    if (half_out) ((_Float16*)Cout)[idx] = (_Float16)v;
    else          ((float*)Cout)[idx] = v;
  }
}

// ---------------------------------------------------------------------------
// Fused LSTM step: pre = xp[:,t,:] + h_in @ Whh^T ; gates ; update c ; emit h.
// grid = (H/16, B/16), block = 128 (4 waves, one per gate).
// h-state lives in f16; tile staged to LDS via async global->LDS b128.
// ---------------------------------------------------------------------------
__global__ void lstm_step_kernel(const float* __restrict__ xp, const _Float16* __restrict__ Whh,
                                 const _Float16* __restrict__ h16_in, float* __restrict__ c_st,
                                 _Float16* __restrict__ h16_out, float* __restrict__ enc,
                                 _Float16* __restrict__ enc16,
                                 int t_x, int t_store, int enc_off) {
  __shared__ __align__(16) _Float16 As[16 * HH];      // 8 KB
  __shared__ float gbuf[4][16][16];
  int tid = threadIdx.x;
  int m0 = blockIdx.y * 16;  // batch tile
  int h0 = blockIdx.x * 16;  // hidden-col tile
  // h tile rows are full contiguous rows -> one contiguous 8KB block.
  {
    unsigned lbase = (unsigned)(size_t)(&As[0]);
    const char* src = (const char*)(h16_in + (size_t)m0 * HH);
    for (int idx = tid; idx < (16 * HH) / 8; idx += 128)
      async_copy_b128(lbase + idx * 16, src + idx * 16);
    wait_async_all();
  }
  __syncthreads();
  int wave = tid >> 5, lane = tid & 31;
  int n0g = wave * HH + h0;  // column in the 4H-wide gate gemm
  fv8 acc = {};
  for (int k0 = 0; k0 < HH; k0 += 32) {
    hv16 a = frag_a_h(As, HH, 0, k0, lane);
    hv16 b = frag_b_wT(Whh, HH, n0g, k0, lane);
    acc = WMMA(a, b, acc);
  }
  int n = lane & 15, hl = lane >> 4;
  for (int r = 0; r < 8; ++r) gbuf[wave][r + 8 * hl][n] = acc[r];
  __syncthreads();
  for (int idx = tid; idx < 256; idx += 128) {
    int m = idx >> 4, nn = idx & 15;
    int b = m0 + m, hc = h0 + nn;
    const float* xr = xp + ((size_t)b * SS + t_x) * (4 * HH);
    float iv = sigm(gbuf[0][m][nn] + xr[0 * HH + hc]);
    float fv = sigm(gbuf[1][m][nn] + xr[1 * HH + hc]);
    float gv = tanhf(gbuf[2][m][nn] + xr[2 * HH + hc]);
    float ov = sigm(gbuf[3][m][nn] + xr[3 * HH + hc]);
    size_t sidx = (size_t)b * HH + hc;
    float cv = fv * c_st[sidx] + iv * gv;
    float hv = ov * tanhf(cv);
    c_st[sidx] = cv;
    h16_out[sidx] = (_Float16)hv;
    size_t eidx = ((size_t)b * SS + t_store) * (2 * HH) + enc_off + hc;
    enc[eidx] = hv;
    enc16[eidx] = (_Float16)hv;
  }
}

// ---------------------------------------------------------------------------
// Fused decoder cell: pre = din @ Wih^T + h_in @ Whh^T + bias ; update c,h.
// grid = (H/16, B/16), block = 128. Both A tiles staged via async LDS copy.
// ---------------------------------------------------------------------------
__global__ void dec_cell_kernel(const _Float16* __restrict__ din16, const _Float16* __restrict__ Wih,
                                const _Float16* __restrict__ h16_in, const _Float16* __restrict__ Whh,
                                const float* __restrict__ bias, float* __restrict__ c_st,
                                _Float16* __restrict__ h16_out) {
  __shared__ __align__(16) _Float16 As1[16 * 2 * HH];  // 16 KB
  __shared__ __align__(16) _Float16 As2[16 * HH];      // 8 KB
  __shared__ float gbuf[4][16][16];
  int tid = threadIdx.x;
  int m0 = blockIdx.y * 16;
  int h0 = blockIdx.x * 16;
  {
    unsigned l1 = (unsigned)(size_t)(&As1[0]);
    const char* s1 = (const char*)(din16 + (size_t)m0 * (2 * HH));
    for (int idx = tid; idx < (16 * 2 * HH) / 8; idx += 128)
      async_copy_b128(l1 + idx * 16, s1 + idx * 16);
    unsigned l2 = (unsigned)(size_t)(&As2[0]);
    const char* s2 = (const char*)(h16_in + (size_t)m0 * HH);
    for (int idx = tid; idx < (16 * HH) / 8; idx += 128)
      async_copy_b128(l2 + idx * 16, s2 + idx * 16);
    wait_async_all();
  }
  __syncthreads();
  int wave = tid >> 5, lane = tid & 31;
  int n0g = wave * HH + h0;
  fv8 acc = {};
  for (int k0 = 0; k0 < 2 * HH; k0 += 32) {
    hv16 a = frag_a_h(As1, 2 * HH, 0, k0, lane);
    hv16 b = frag_b_wT(Wih, 2 * HH, n0g, k0, lane);
    acc = WMMA(a, b, acc);
  }
  for (int k0 = 0; k0 < HH; k0 += 32) {
    hv16 a = frag_a_h(As2, HH, 0, k0, lane);
    hv16 b = frag_b_wT(Whh, HH, n0g, k0, lane);
    acc = WMMA(a, b, acc);
  }
  int n = lane & 15, hl = lane >> 4;
  float bv = bias[n0g + n];
  for (int r = 0; r < 8; ++r) gbuf[wave][r + 8 * hl][n] = acc[r] + bv;
  __syncthreads();
  for (int idx = tid; idx < 256; idx += 128) {
    int m = idx >> 4, nn = idx & 15;
    int b = m0 + m, hc = h0 + nn;
    float iv = sigm(gbuf[0][m][nn]);
    float fv = sigm(gbuf[1][m][nn]);
    float gv = tanhf(gbuf[2][m][nn]);
    float ov = sigm(gbuf[3][m][nn]);
    size_t sidx = (size_t)b * HH + hc;
    float cv = fv * c_st[sidx] + iv * gv;
    c_st[sidx] = cv;
    h16_out[sidx] = (_Float16)(ov * tanhf(cv));
  }
}

// ---------------------------------------------------------------------------
// Embed: emb16 = f16(layernorm(tanh(x @ proj_W^T + proj_b))). grid=B*S, blk=D.
// ---------------------------------------------------------------------------
__global__ void embed_kernel(const float* __restrict__ x, const float* __restrict__ pW,
                             const float* __restrict__ pb, const float* __restrict__ lng,
                             const float* __restrict__ lnb, _Float16* __restrict__ emb16) {
  __shared__ float red[DD];
  int row = blockIdx.x;
  int d = threadIdx.x;
  const float* xr = x + (size_t)row * IN;
  float acc = pb[d];
#pragma unroll
  for (int k = 0; k < IN; ++k) acc += xr[k] * pW[d * IN + k];
  float v = tanhf(acc);
  red[d] = v;
  __syncthreads();
  for (int o = DD / 2; o > 0; o >>= 1) { if (d < o) red[d] += red[d + o]; __syncthreads(); }
  float mean = red[0] * (1.0f / DD);
  __syncthreads();
  float dv = v - mean;
  red[d] = dv * dv;
  __syncthreads();
  for (int o = DD / 2; o > 0; o >>= 1) { if (d < o) red[d] += red[d + o]; __syncthreads(); }
  float var = red[0] * (1.0f / DD);
  emb16[(size_t)row * DD + d] = (_Float16)(dv * rsqrtf(var + 1e-5f) * lng[d] + lnb[d]);
}

// scores[b,s] = scale * sum_h tanh(ref[b,s,h] + q[b,h]) * v[h].  4 waves/block.
__global__ void attn_score_kernel(const float* __restrict__ ref, const float* __restrict__ q,
                                  const float* __restrict__ v, float* __restrict__ out,
                                  float scale) {
  int row = blockIdx.x * 4 + (threadIdx.x >> 5);  // b*S + s
  int lane = threadIdx.x & 31;
  int b = row >> 7;
  const float* rp = ref + (size_t)row * HH;
  const float* qp = q + (size_t)b * HH;
  float sum = 0.0f;
  for (int h = lane; h < HH; h += 32) sum += tanhf(rp[h] + qp[h]) * v[h];
  for (int o = 16; o > 0; o >>= 1) sum += __shfl_xor(sum, o, 32);
  if (lane == 0) out[row] = sum * scale;
}

// Masked softmax over s, ctx = gp @ enc (first H cols), hpc16 = f16(h + ctx).
__global__ void softmax_ctx_kernel(const float* __restrict__ gs, const int* __restrict__ mask,
                                   const float* __restrict__ enc, const _Float16* __restrict__ h16,
                                   _Float16* __restrict__ hpc16) {
  __shared__ float p[SS];
  __shared__ float red[SS];
  int b = blockIdx.x, s = threadIdx.x;
  float val = mask[b * SS + s] ? -1e9f : gs[b * SS + s];
  red[s] = val;
  __syncthreads();
  for (int o = SS / 2; o > 0; o >>= 1) { if (s < o) red[s] = fmaxf(red[s], red[s + o]); __syncthreads(); }
  float mx = red[0];
  __syncthreads();
  float e = __expf(val - mx);
  red[s] = e;
  __syncthreads();
  for (int o = SS / 2; o > 0; o >>= 1) { if (s < o) red[s] += red[s + o]; __syncthreads(); }
  p[s] = e / red[0];
  __syncthreads();
  for (int j = s; j < HH; j += SS) {
    float a = 0.0f;
    for (int ss = 0; ss < SS; ++ss) a += p[ss] * enc[((size_t)b * SS + ss) * (2 * HH) + j];
    hpc16[(size_t)b * HH + j] = (_Float16)((float)h16[(size_t)b * HH + j] + a);
  }
}

// ps -> masked logits, argmax -> sel, update mask, gather din16 = enc16[b,sel].
__global__ void ps_select_kernel(const float* __restrict__ pref, const float* __restrict__ hq,
                                 const float* __restrict__ pv, int* __restrict__ mask,
                                 const _Float16* __restrict__ enc16, float* __restrict__ logits,
                                 float* __restrict__ ptrs, _Float16* __restrict__ din16,
                                 float scale, int t) {
  __shared__ float red[SS];
  __shared__ int ridx[SS];
  __shared__ int sel_s;
  int b = blockIdx.x, s = threadIdx.x;
  const float* rp = pref + ((size_t)b * SS + s) * HH;
  const float* qp = hq + (size_t)b * HH;
  float sum = 0.0f;
  for (int h = 0; h < HH; ++h) sum += tanhf(rp[h] + qp[h]) * pv[h];
  float ps = mask[b * SS + s] ? -1e9f : sum * scale;
  logits[((size_t)b * SS + t) * SS + s] = ps;
  red[s] = ps;
  ridx[s] = s;
  __syncthreads();
  for (int o = SS / 2; o > 0; o >>= 1) {
    if (s < o && red[s + o] > red[s]) { red[s] = red[s + o]; ridx[s] = ridx[s + o]; }
    __syncthreads();
  }
  if (s == 0) {
    int sel = ridx[0];
    sel_s = sel;
    mask[b * SS + sel] = 1;
    ptrs[(size_t)b * SS + t] = (float)sel;
  }
  __syncthreads();
  int sel = sel_s;
  for (int j = s; j < 2 * HH; j += SS)
    din16[(size_t)b * (2 * HH) + j] = enc16[((size_t)b * SS + sel) * (2 * HH) + j];
}

__global__ void f32_to_f16_kernel(const float* __restrict__ s, _Float16* __restrict__ d, int n) {
  int i = blockIdx.x * blockDim.x + threadIdx.x;
  if (i < n) d[i] = (_Float16)s[i];
}

__global__ void add_vec_kernel(const float* __restrict__ a, const float* __restrict__ b,
                               float* __restrict__ c, int n) {
  int i = blockIdx.x * blockDim.x + threadIdx.x;
  if (i < n) c[i] = a[i] + b[i];
}

// concat copies into f16 buffers
__global__ void copy_slice_h2h_kernel(const _Float16* __restrict__ src, _Float16* __restrict__ dst,
                                      int cols, int ldd, int off) {
  int b = blockIdx.x;
  for (int j = threadIdx.x; j < cols; j += blockDim.x)
    dst[(size_t)b * ldd + off + j] = src[(size_t)b * cols + j];
}
__global__ void copy_slice_f2h_kernel(const float* __restrict__ src, _Float16* __restrict__ dst,
                                      int cols, int ldd, int off) {
  int b = blockIdx.x;
  for (int j = threadIdx.x; j < cols; j += blockDim.x)
    dst[(size_t)b * ldd + off + j] = (_Float16)src[(size_t)b * cols + j];
}

__global__ void mean_enc_kernel(const float* __restrict__ enc, _Float16* __restrict__ din16) {
  int b = blockIdx.x;
  for (int j = threadIdx.x; j < 2 * HH; j += blockDim.x) {
    float s = 0.0f;
    for (int t = 0; t < SS; ++t) s += enc[((size_t)b * SS + t) * (2 * HH) + j];
    din16[(size_t)b * (2 * HH) + j] = (_Float16)(s * (1.0f / SS));
  }
}

// ---------------------------------------------------------------------------
extern "C" void kernel_launch(void* const* d_in, const int* in_sizes, int n_in,
                              void* d_out, int out_size, void* d_ws, size_t ws_size,
                              hipStream_t stream) {
  (void)in_sizes; (void)n_in; (void)out_size; (void)ws_size;
  const float* x      = (const float*)d_in[0];
  const float* proj_W = (const float*)d_in[1];
  const float* proj_b = (const float*)d_in[2];
  const float* ln_g   = (const float*)d_in[3];
  const float* ln_b   = (const float*)d_in[4];
  const float* Wih_f  = (const float*)d_in[5];
  const float* Whh_f  = (const float*)d_in[6];
  const float* bih_f  = (const float*)d_in[7];
  const float* bhh_f  = (const float*)d_in[8];
  const float* Wih_b  = (const float*)d_in[9];
  const float* Whh_b  = (const float*)d_in[10];
  const float* bih_b  = (const float*)d_in[11];
  const float* bhh_b  = (const float*)d_in[12];
  const float* Wih_d  = (const float*)d_in[13];
  const float* Whh_d  = (const float*)d_in[14];
  const float* bih_d  = (const float*)d_in[15];
  const float* bhh_d  = (const float*)d_in[16];
  const float* hb_W   = (const float*)d_in[17];
  const float* hb_b   = (const float*)d_in[18];
  const float* cb_W   = (const float*)d_in[19];
  const float* cb_b   = (const float*)d_in[20];
  const float* gW1    = (const float*)d_in[21];
  const float* gW2    = (const float*)d_in[22];
  const float* gv     = (const float*)d_in[23];
  const float* pW1    = (const float*)d_in[24];
  const float* pW2    = (const float*)d_in[25];
  const float* pv     = (const float*)d_in[26];

  // ---- workspace bump allocator (256B-aligned slices) ----
  char* base = (char*)d_ws;
  size_t off = 0;
  auto alloc = [&](size_t elems, size_t esz) -> void* {
    void* p = base + off;
    off += (elems * esz + 255) & ~(size_t)255;
    return p;
  };
  _Float16* emb16 = (_Float16*)alloc((size_t)BB * SS * DD, 2);
  float* xpF  = (float*)alloc((size_t)BB * SS * 4 * HH, 4);
  float* xpB  = (float*)alloc((size_t)BB * SS * 4 * HH, 4);
  float* enc  = (float*)alloc((size_t)BB * SS * 2 * HH, 4);
  _Float16* enc16 = (_Float16*)alloc((size_t)BB * SS * 2 * HH, 2);
  float* gref = (float*)alloc((size_t)BB * SS * HH, 4);
  float* pref = (float*)alloc((size_t)BB * SS * HH, 4);
  _Float16* h16A = (_Float16*)alloc((size_t)BB * HH, 2);
  _Float16* h16B = (_Float16*)alloc((size_t)BB * HH, 2);
  float* cS   = (float*)alloc((size_t)BB * HH, 4);
  _Float16* hcat16 = (_Float16*)alloc((size_t)BB * 2 * HH, 2);
  _Float16* ccat16 = (_Float16*)alloc((size_t)BB * 2 * HH, 2);
  _Float16* din16  = (_Float16*)alloc((size_t)BB * 2 * HH, 2);
  float* qb   = (float*)alloc((size_t)BB * HH, 4);
  float* hq   = (float*)alloc((size_t)BB * HH, 4);
  _Float16* hpc16 = (_Float16*)alloc((size_t)BB * HH, 2);
  float* gsb  = (float*)alloc((size_t)BB * SS, 4);
  int*   mask = (int*)alloc((size_t)BB * SS, 4);
  float* biasF = (float*)alloc(4 * HH, 4);
  float* biasB = (float*)alloc(4 * HH, 4);
  float* biasD = (float*)alloc(4 * HH, 4);
  _Float16* WihF16 = (_Float16*)alloc((size_t)4 * HH * DD, 2);
  _Float16* WhhF16 = (_Float16*)alloc((size_t)4 * HH * HH, 2);
  _Float16* WihB16 = (_Float16*)alloc((size_t)4 * HH * DD, 2);
  _Float16* WhhB16 = (_Float16*)alloc((size_t)4 * HH * HH, 2);
  _Float16* WihD16 = (_Float16*)alloc((size_t)4 * HH * 2 * HH, 2);
  _Float16* WhhD16 = (_Float16*)alloc((size_t)4 * HH * HH, 2);
  _Float16* hbW16  = (_Float16*)alloc((size_t)HH * 2 * HH, 2);
  _Float16* cbW16  = (_Float16*)alloc((size_t)HH * 2 * HH, 2);
  _Float16* gW1_16 = (_Float16*)alloc((size_t)HH * 2 * HH, 2);
  _Float16* gW2_16 = (_Float16*)alloc((size_t)HH * HH, 2);
  _Float16* pW1_16 = (_Float16*)alloc((size_t)HH * 2 * HH, 2);
  _Float16* pW2_16 = (_Float16*)alloc((size_t)HH * HH, 2);

  auto conv = [&](const float* s, _Float16* d, int n) {
    f32_to_f16_kernel<<<dim3((n + 255) / 256), dim3(256), 0, stream>>>(s, d, n);
  };
  conv(Wih_f, WihF16, 4 * HH * DD);
  conv(Whh_f, WhhF16, 4 * HH * HH);
  conv(Wih_b, WihB16, 4 * HH * DD);
  conv(Whh_b, WhhB16, 4 * HH * HH);
  conv(Wih_d, WihD16, 4 * HH * 2 * HH);
  conv(Whh_d, WhhD16, 4 * HH * HH);
  conv(hb_W, hbW16, HH * 2 * HH);
  conv(cb_W, cbW16, HH * 2 * HH);
  conv(gW1, gW1_16, HH * 2 * HH);
  conv(gW2, gW2_16, HH * HH);
  conv(pW1, pW1_16, HH * 2 * HH);
  conv(pW2, pW2_16, HH * HH);
  add_vec_kernel<<<dim3(4), dim3(256), 0, stream>>>(bih_f, bhh_f, biasF, 4 * HH);
  add_vec_kernel<<<dim3(4), dim3(256), 0, stream>>>(bih_b, bhh_b, biasB, 4 * HH);
  add_vec_kernel<<<dim3(4), dim3(256), 0, stream>>>(bih_d, bhh_d, biasD, 4 * HH);

  // 1) embedding + layernorm (emitted directly as f16 WMMA operand)
  embed_kernel<<<dim3(BB * SS), dim3(DD), 0, stream>>>(x, proj_W, proj_b, ln_g, ln_b, emb16);

  // 2) xp = emb @ Wih^T + (bih + bhh)   [16384 x 256 -> 1024], f32 out
  gemm_wmma_kernel<<<dim3(4 * HH / 16, BB * SS / 16), dim3(32), 0, stream>>>(
      emb16, WihF16, biasF, xpF, DD, DD, DD, 4 * HH, 0, 0);
  gemm_wmma_kernel<<<dim3(4 * HH / 16, BB * SS / 16), dim3(32), 0, stream>>>(
      emb16, WihB16, biasB, xpB, DD, DD, DD, 4 * HH, 0, 0);

  // 3) forward LSTM scan
  hipMemsetAsync(h16A, 0, (size_t)BB * HH * 2, stream);
  hipMemsetAsync(cS, 0, (size_t)BB * HH * 4, stream);
  _Float16* hc = h16A; _Float16* hn = h16B;
  for (int t = 0; t < SS; ++t) {
    lstm_step_kernel<<<dim3(HH / 16, BB / 16), dim3(128), 0, stream>>>(
        xpF, WhhF16, hc, cS, hn, enc, enc16, t, t, 0);
    _Float16* tmp = hc; hc = hn; hn = tmp;
  }
  copy_slice_h2h_kernel<<<dim3(BB), dim3(128), 0, stream>>>(hc, hcat16, HH, 2 * HH, 0);
  copy_slice_f2h_kernel<<<dim3(BB), dim3(128), 0, stream>>>(cS, ccat16, HH, 2 * HH, 0);

  // 4) backward LSTM scan (reversed time, written back reversed)
  hipMemsetAsync(h16A, 0, (size_t)BB * HH * 2, stream);
  hipMemsetAsync(cS, 0, (size_t)BB * HH * 4, stream);
  hc = h16A; hn = h16B;
  for (int tau = 0; tau < SS; ++tau) {
    int t = SS - 1 - tau;
    lstm_step_kernel<<<dim3(HH / 16, BB / 16), dim3(128), 0, stream>>>(
        xpB, WhhB16, hc, cS, hn, enc, enc16, t, t, HH);
    _Float16* tmp = hc; hc = hn; hn = tmp;
  }
  copy_slice_h2h_kernel<<<dim3(BB), dim3(128), 0, stream>>>(hc, hcat16, HH, 2 * HH, HH);
  copy_slice_f2h_kernel<<<dim3(BB), dim3(128), 0, stream>>>(cS, ccat16, HH, 2 * HH, HH);

  // 5) attention reference projections: g_ref/p_ref = enc @ {gW1,pW1}^T (f32 out)
  gemm_wmma_kernel<<<dim3(HH / 16, BB * SS / 16), dim3(32), 0, stream>>>(
      enc16, gW1_16, nullptr, gref, 2 * HH, 2 * HH, 2 * HH, HH, 0, 0);
  gemm_wmma_kernel<<<dim3(HH / 16, BB * SS / 16), dim3(32), 0, stream>>>(
      enc16, pW1_16, nullptr, pref, 2 * HH, 2 * HH, 2 * HH, HH, 0, 0);

  // 6) decoder init: h16 = f16(tanh(hcat @ hb_W^T + hb_b)), c = tanh(... cb ...)
  gemm_wmma_kernel<<<dim3(HH / 16, BB / 16), dim3(32), 0, stream>>>(
      hcat16, hbW16, hb_b, h16A, 2 * HH, 2 * HH, 2 * HH, HH, 1, 1);
  gemm_wmma_kernel<<<dim3(HH / 16, BB / 16), dim3(32), 0, stream>>>(
      ccat16, cbW16, cb_b, cS, 2 * HH, 2 * HH, 2 * HH, HH, 1, 0);
  mean_enc_kernel<<<dim3(BB), dim3(128), 0, stream>>>(enc, din16);
  hipMemsetAsync(mask, 0, (size_t)BB * SS * 4, stream);

  // 7) decoder loop
  float* logits = (float*)d_out;
  float* ptrs = logits + (size_t)BB * SS * SS;
  const float scale = 0.0625f;  // 1/sqrt(256)
  hc = h16A; hn = h16B;
  for (int t = 0; t < SS; ++t) {
    dec_cell_kernel<<<dim3(HH / 16, BB / 16), dim3(128), 0, stream>>>(
        din16, WihD16, hc, WhhD16, biasD, cS, hn);
    gemm_wmma_kernel<<<dim3(HH / 16, BB / 16), dim3(32), 0, stream>>>(
        hn, gW2_16, nullptr, qb, HH, HH, HH, HH, 0, 0);
    attn_score_kernel<<<dim3(BB * SS / 4), dim3(128), 0, stream>>>(gref, qb, gv, gsb, scale);
    softmax_ctx_kernel<<<dim3(BB), dim3(SS), 0, stream>>>(gsb, mask, enc, hn, hpc16);
    gemm_wmma_kernel<<<dim3(HH / 16, BB / 16), dim3(32), 0, stream>>>(
        hpc16, pW2_16, nullptr, hq, HH, HH, HH, HH, 0, 0);
    ps_select_kernel<<<dim3(BB), dim3(SS), 0, stream>>>(
        pref, hq, pv, mask, enc16, logits, ptrs, din16, scale, t);
    _Float16* tmp = hc; hc = hn; hn = tmp;
  }
}
